// DTGCN_37297495998604
// MI455X (gfx1250) — matrix operand order repaired and checked by reading, not compiled, not run on hardware
//
#include <hip/hip_runtime.h>
#include <hip/hip_bf16.h>

typedef __attribute__((ext_vector_type(16))) _Float16 v16h;
typedef __attribute__((ext_vector_type(8)))  _Float16 v8h;
typedef __attribute__((ext_vector_type(8)))  float    v8f;

namespace {
constexpr int B_ = 16, N_ = 256, T_ = 64, W_ = 12, H_ = 64, ED_ = 64, S_ = 5, C_ = 4, E_ = 2048;
constexpr int FG_  = W_ + H_;     // 76
constexpr int KIT_ = 96;          // It GEMM K padded to multiple of 32
constexpr int MALL = B_ * N_;     // 4096
}

// ---------------------------------------------------------------- utilities
__global__ void zero_f32(float* p, int n) {
    int i = blockIdx.x * blockDim.x + threadIdx.x;
    int stride = gridDim.x * blockDim.x;
    for (; i < n; i += stride) p[i] = 0.f;
}

__global__ void f32_to_f16(const float* __restrict__ s, _Float16* __restrict__ d, int n) {
    int i = blockIdx.x * blockDim.x + threadIdx.x;
    if (i < n) d[i] = (_Float16)s[i];
}

// --------------------------------------------------- one-time static setup
// All GEMM B operands are stored TRANSPOSED (Bt row-major [Nn x K]) so the
// WMMA B-fragment per lane is 16 contiguous halves -> one b128-pair load.
__global__ void setup_static(const float* __restrict__ edge_w, const int* __restrict__ edge_idx,
                             const float* __restrict__ gconv_w,
                             const float* __restrict__ w1, const float* __restrict__ w2,
                             const float* __restrict__ wz_c, const float* __restrict__ bz_c,
                             const float* __restrict__ wz_l, const float* __restrict__ bz_l,
                             const float* __restrict__ wr_c, const float* __restrict__ br_c,
                             const float* __restrict__ wr_l, const float* __restrict__ br_l,
                             const float* __restrict__ wh_c, const float* __restrict__ bh_c,
                             const float* __restrict__ wh_l, const float* __restrict__ bh_l,
                             float* A_stat, float* dinv_stat, _Float16* AnT_h,
                             _Float16* gwT_h, _Float16* w1T_h, _Float16* w2T_h,
                             _Float16* wzlT_h, _Float16* wrlT_h, _Float16* whlT_h,
                             float* pq /* 6 x 64 : pz qz pr qr ph qh */)
{
    const int tid = threadIdx.x, nt = blockDim.x;           // 256 threads
    for (int i = tid; i < N_ * N_; i += nt) A_stat[i] = 0.f;
    __syncthreads();
    for (int e = tid; e < E_; e += nt) {
        int s = edge_idx[e], d = edge_idx[E_ + e];
        atomicAdd(&A_stat[s * N_ + d], edge_w[e]);
    }
    __syncthreads();
    if (tid < N_) {                                         // deg over rows i (axis -2), +1 self loop
        float deg = 1.f;
        for (int i = 0; i < N_; ++i) deg += A_stat[i * N_ + tid];
        dinv_stat[tid] = rsqrtf(deg);
    }
    __syncthreads();
    for (int idx = tid; idx < N_ * N_; idx += nt) {         // AnT[j][i] = dinv[i]*Ahat[i][j]*dinv[j]
        int j = idx / N_, i = idx - j * N_;
        float v = A_stat[i * N_ + j] + (i == j ? 1.f : 0.f);
        AnT_h[idx] = (_Float16)(dinv_stat[i] * v * dinv_stat[j]);
    }
    for (int idx = tid; idx < ED_ * KIT_; idx += nt) {      // gwT[f][k] (zero-padded K)
        int f = idx / KIT_, k = idx - f * KIT_;
        gwT_h[idx] = (_Float16)(k < FG_ ? gconv_w[k * ED_ + f] : 0.f);
    }
    for (int idx = tid; idx < ED_ * ED_; idx += nt) {       // Bt[f][g] = B[g][f]
        int f = idx / H_, g = idx - f * H_;
        w1T_h[idx]  = (_Float16)w1[g * H_ + f];
        w2T_h[idx]  = (_Float16)w2[g * H_ + f];
        wzlT_h[idx] = (_Float16)wz_l[(H_ + g) * H_ + f];    // bottom half of concat weight
        wrlT_h[idx] = (_Float16)wr_l[(H_ + g) * H_ + f];
        whlT_h[idx] = (_Float16)wh_l[(H_ + g) * H_ + f];
    }
    if (tid < H_) {                                         // rank-1 collapse of c-channel matmuls
        float pz = 0, qz = 0, pr = 0, qr = 0, ph = 0, qh = 0;
        for (int g = 0; g < H_; ++g) {
            pz += wz_c[g] * wz_l[g * H_ + tid];  qz += bz_c[g] * wz_l[g * H_ + tid];
            pr += wr_c[g] * wr_l[g * H_ + tid];  qr += br_c[g] * wr_l[g * H_ + tid];
            ph += wh_c[g] * wh_l[g * H_ + tid];  qh += bh_c[g] * wh_l[g * H_ + tid];
        }
        pq[0 * H_ + tid] = pz;  pq[1 * H_ + tid] = qz + bz_l[tid];
        pq[2 * H_ + tid] = pr;  pq[3 * H_ + tid] = qr + br_l[tid];
        pq[4 * H_ + tid] = ph;  pq[5 * H_ + tid] = qh + bh_l[tid];
    }
}

// ------------------------------------------------- generic batched f16 GEMM
// C[M x Nn] = A[M x K] * B[K x Nn] with B supplied TRANSPOSED:
//   A  row-major [M x K]  (lda = K)
//   Bt row-major [Nn x K] (ldb = K)   -> per-lane fragment is contiguous
//   C  row-major [M x Nn]
// f32 accumulation via V_WMMA_F32_16X16X32_F16. Block = 4 waves, 64x64 C tile.
// Requires M%64==0, Nn%64==0, K%32==0. No divergence: EXEC stays all-ones.
__global__ __launch_bounds__(128) void gemm_f16_wmma(
    const _Float16* __restrict__ A, const _Float16* __restrict__ Bt,
    float* __restrict__ Cm, int M, int Nn, int K,
    long sA, long sB, long sC)
{
    A  += (long)blockIdx.z * sA;
    Bt += (long)blockIdx.z * sB;
    Cm += (long)blockIdx.z * sC;
    const int lane = threadIdx.x & 31;
    const int wave = threadIdx.x >> 5;
    const int m0   = blockIdx.y * 64 + wave * 16;
    const int n0   = blockIdx.x * 64;
    const int half = lane >> 4;          // K-half select (A/B), M-half select (C)
    const int l16  = lane & 15;          // A: row M ; B/C: col N

    const _Float16* Arow = A + (long)(m0 + l16) * K + half * 16;         // 32B-aligned chunks
    const _Float16* Brow[4];
    #pragma unroll
    for (int ntile = 0; ntile < 4; ++ntile)
        Brow[ntile] = Bt + (long)(n0 + ntile * 16 + l16) * K + half * 16; // 32B-aligned

    v8f acc[4] = {v8f{}, v8f{}, v8f{}, v8f{}};
    for (int k0 = 0; k0 < K; k0 += 32) {
        // A fragment: a[j] <- k = k0+half*16+j, contiguous (matches B)
        v8h alo = *(const v8h*)(Arow + k0);
        v8h ahi = *(const v8h*)(Arow + k0 + 8);
        v16h a = __builtin_shufflevector(alo, ahi, 0, 1, 2, 3, 4, 5, 6, 7,
                                         8, 9, 10, 11, 12, 13, 14, 15);
        if (k0 + 32 < K)                                   // uniform cond: EXEC stays full
            __builtin_prefetch(Arow + k0 + 32, 0, 1);      // -> global_prefetch_b8
        #pragma unroll
        for (int ntile = 0; ntile < 4; ++ntile) {
            // B fragment: b[j] = B[k0+16*half+j][n] = Bt[n][k0+16*half+j], contiguous
            v16h b = *(const v16h*)(Brow[ntile] + k0);
            acc[ntile] = __builtin_amdgcn_wmma_f32_16x16x32_f16(
                false, a, false, b, (short)0, acc[ntile], false, false);
        }
    }
    // C fragment: VGPR r -> row m0 + r + 8*half, col n0 + ntile*16 + l16
    #pragma unroll
    for (int ntile = 0; ntile < 4; ++ntile)
        #pragma unroll
        for (int r = 0; r < 8; ++r)
            Cm[(long)(m0 + r + half * 8) * Nn + n0 + ntile * 16 + l16] = acc[ntile][r];
}

// ------------------------------------------------------ per-step elementwise
__global__ void build_it(const float* __restrict__ x, const float* __restrict__ h,
                         _Float16* __restrict__ It, int t)
{
    int idx = blockIdx.x * blockDim.x + threadIdx.x;
    if (idx >= MALL * KIT_) return;
    int m = idx / KIT_, k = idx - m * KIT_;
    float v = 0.f;
    if (k < W_) {
        int tau = t - (W_ - 1) + k;                // window ending at t, left zero-pad
        if (tau >= 0) v = x[m * T_ + tau];
    } else if (k < FG_) {
        v = h[m * H_ + (k - W_)];
    }
    It[idx] = (_Float16)v;
}

// Y[4096 x 64] -> per-batch transposed f16: YT[b][f][i]  (Bt layout for df GEMM)
__global__ void y_to_bt(const float* __restrict__ Y, _Float16* __restrict__ YT)
{
    int idx = blockIdx.x * blockDim.x + threadIdx.x;
    if (idx >= MALL * ED_) return;
    int m = idx / ED_, f = idx - m * ED_;
    int b = m >> 8, i = m & 255;
    YT[((long)b * ED_ + f) * N_ + i] = (_Float16)Y[idx];
}

__global__ void bias_to_f16(const float* __restrict__ s, const float* __restrict__ bias,
                            _Float16* __restrict__ d, int n)
{
    int i = blockIdx.x * blockDim.x + threadIdx.x;
    if (i < n) d[i] = (_Float16)(s[i] + bias[i & (ED_ - 1)]);
}

__global__ void epi_de(const float* __restrict__ E1, const float* __restrict__ E2,
                       const float* __restrict__ b1, const float* __restrict__ b2,
                       _Float16* __restrict__ de1, _Float16* __restrict__ de2)
{
    int idx = blockIdx.x * blockDim.x + threadIdx.x;
    if (idx >= MALL * ED_) return;
    int f = idx & (ED_ - 1);
    de1[idx] = (_Float16)tanhf(E1[idx] + b1[f]);
    de2[idx] = (_Float16)tanhf(E2[idx] + b2[f]);   // row-major de2 IS Bt for de1 @ de2^T
}

__global__ void epi_et(const float* __restrict__ M12, float* __restrict__ Eh, int slot)
{
    int idx = blockIdx.x * blockDim.x + threadIdx.x;
    if (idx >= B_ * N_ * N_) return;
    int b = idx >> 16, rest = idx & 65535;
    int i = rest >> 8, j = rest & 255;
    float e = tanhf(M12[idx] - M12[(b << 16) + (j << 8) + i]);
    Eh[(((long)b * S_ + slot) << 16) + rest] = e > 0.f ? e : 0.f;
}

__global__ void deg_dinv(const float* __restrict__ Eh, const float* __restrict__ x,
                         float* __restrict__ dinv, float* __restrict__ u,
                         int t, float inv_cnt)
{
    int b = blockIdx.x, j = threadIdx.x;                     // grid 16, block 256
    float acc = 0.f;
    for (int s = 0; s < S_; ++s) {
        const float* p = Eh + (((long)b * S_ + s) << 16) + j;
        for (int i = 0; i < N_; ++i) acc += p[i << 8];       // column sum, coalesced over j
    }
    float dv = rsqrtf(1.f + acc * inv_cnt);                  // deg >= 1 always (relu edges)
    dinv[b * N_ + j] = dv;
    u[b * N_ + j] = dv * x[(b * N_ + j) * T_ + t];
}

__global__ void compute_avec(const float* __restrict__ Eh, const float* __restrict__ dinv,
                             const float* __restrict__ u, float* __restrict__ avec, float inv_cnt)
{
    int b = blockIdx.x, j = threadIdx.x;
    float acc = 0.f;
    for (int s = 0; s < S_; ++s) {
        const float* p = Eh + (((long)b * S_ + s) << 16) + j;
        for (int i = 0; i < N_; ++i) acc += p[i << 8] * u[b * N_ + i];
    }
    avec[b * N_ + j] = dinv[b * N_ + j] * (u[b * N_ + j] + acc * inv_cnt);
}

__global__ void epi_zr(const float* __restrict__ avec, const float* __restrict__ pq,
                       const float* __restrict__ Gz, const float* __restrict__ Gr,
                       const float* __restrict__ h, float* __restrict__ zb,
                       _Float16* __restrict__ hr)
{
    int idx = blockIdx.x * blockDim.x + threadIdx.x;
    if (idx >= MALL * H_) return;
    int m = idx / H_, f = idx - m * H_;
    float a = avec[m];
    float z = 1.f / (1.f + expf(-(a * pq[0 * H_ + f] + pq[1 * H_ + f] + Gz[idx])));
    float r = 1.f / (1.f + expf(-(a * pq[2 * H_ + f] + pq[3 * H_ + f] + Gr[idx])));
    zb[idx] = z;
    hr[idx] = (_Float16)(h[idx] * r);
}

__global__ void epi_update_h(const float* __restrict__ avec, const float* __restrict__ pq,
                             const float* __restrict__ Gh, const float* __restrict__ zb,
                             float* __restrict__ h)
{
    int idx = blockIdx.x * blockDim.x + threadIdx.x;
    if (idx >= MALL * H_) return;
    int m = idx / H_, f = idx - m * H_;
    float ht = tanhf(avec[m] * pq[4 * H_ + f] + pq[5 * H_ + f] + Gh[idx]);
    float z = zb[idx];
    h[idx] = z * h[idx] + (1.f - z) * ht;
}

__global__ void classifier(const float* __restrict__ h, const float* __restrict__ cls_w,
                           const float* __restrict__ cls_b, float* __restrict__ out)
{
    int b = blockIdx.x >> 2, c = blockIdx.x & 3;
    __shared__ float red[256];
    float acc = 0.f;
    for (int k = threadIdx.x; k < N_ * H_; k += 256)
        acc += h[(long)b * N_ * H_ + k] * cls_w[(long)k * C_ + c];
    red[threadIdx.x] = acc;
    __syncthreads();
    for (int s2 = 128; s2 > 0; s2 >>= 1) {
        if (threadIdx.x < s2) red[threadIdx.x] += red[threadIdx.x + s2];
        __syncthreads();
    }
    if (threadIdx.x == 0) out[b * C_ + c] = red[0] + cls_b[c];
}

// ------------------------------------------------------------------- driver
extern "C" void kernel_launch(void* const* d_in, const int* in_sizes, int n_in,
                              void* d_out, int out_size, void* d_ws, size_t ws_size,
                              hipStream_t stream)
{
    (void)in_sizes; (void)n_in; (void)out_size; (void)ws_size;
    const float* x        = (const float*)d_in[0];
    const float* edge_w   = (const float*)d_in[1];
    const float* gconv_w  = (const float*)d_in[2];
    const float* gconv_b  = (const float*)d_in[3];
    const float* w1   = (const float*)d_in[4];
    const float* b1   = (const float*)d_in[5];
    const float* w2   = (const float*)d_in[6];
    const float* b2   = (const float*)d_in[7];
    const float* wz_c = (const float*)d_in[8];
    const float* bz_c = (const float*)d_in[9];
    const float* wz_l = (const float*)d_in[10];
    const float* bz_l = (const float*)d_in[11];
    const float* wr_c = (const float*)d_in[12];
    const float* br_c = (const float*)d_in[13];
    const float* wr_l = (const float*)d_in[14];
    const float* br_l = (const float*)d_in[15];
    const float* wh_c = (const float*)d_in[16];
    const float* bh_c = (const float*)d_in[17];
    const float* wh_l = (const float*)d_in[18];
    const float* bh_l = (const float*)d_in[19];
    const float* cls_w = (const float*)d_in[20];
    const float* cls_b = (const float*)d_in[21];
    const int*   edge_idx = (const int*)d_in[22];

    // ---- workspace carve-out (256B aligned slabs)
    char* wp = (char*)d_ws;
    auto alloc = [&](size_t bytes) -> void* {
        void* p = (void*)wp;
        wp += (bytes + 255) & ~(size_t)255;
        return p;
    };
    float*    A_stat    = (float*)   alloc((size_t)N_ * N_ * 4);
    float*    dinv_stat = (float*)   alloc((size_t)N_ * 4);
    _Float16* AnT_h     = (_Float16*)alloc((size_t)N_ * N_ * 2);
    _Float16* gwT_h     = (_Float16*)alloc((size_t)ED_ * KIT_ * 2);
    _Float16* w1T_h     = (_Float16*)alloc((size_t)ED_ * ED_ * 2);
    _Float16* w2T_h     = (_Float16*)alloc((size_t)ED_ * ED_ * 2);
    _Float16* wzlT_h    = (_Float16*)alloc((size_t)ED_ * ED_ * 2);
    _Float16* wrlT_h    = (_Float16*)alloc((size_t)ED_ * ED_ * 2);
    _Float16* whlT_h    = (_Float16*)alloc((size_t)ED_ * ED_ * 2);
    float*    pq        = (float*)   alloc((size_t)6 * H_ * 4);
    float*    h         = (float*)   alloc((size_t)MALL * H_ * 4);
    float*    Eh        = (float*)   alloc((size_t)B_ * S_ * N_ * N_ * 4);
    _Float16* It_h      = (_Float16*)alloc((size_t)MALL * KIT_ * 2);
    float*    Ybuf      = (float*)   alloc((size_t)MALL * ED_ * 4);
    _Float16* YT_h      = (_Float16*)alloc((size_t)B_ * ED_ * N_ * 2);
    float*    dfbuf     = (float*)   alloc((size_t)MALL * ED_ * 4);
    _Float16* df_h      = (_Float16*)alloc((size_t)MALL * ED_ * 2);
    float*    E1        = (float*)   alloc((size_t)MALL * ED_ * 4);
    float*    E2        = (float*)   alloc((size_t)MALL * ED_ * 4);
    _Float16* de1_h     = (_Float16*)alloc((size_t)MALL * ED_ * 2);
    _Float16* de2_h     = (_Float16*)alloc((size_t)MALL * ED_ * 2);
    float*    M12       = (float*)   alloc((size_t)B_ * N_ * N_ * 4);
    float*    dinv_d    = (float*)   alloc((size_t)B_ * N_ * 4);
    float*    u_d       = (float*)   alloc((size_t)B_ * N_ * 4);
    float*    avec      = (float*)   alloc((size_t)B_ * N_ * 4);
    _Float16* h_h       = (_Float16*)alloc((size_t)MALL * H_ * 2);
    float*    Gz        = (float*)   alloc((size_t)MALL * H_ * 4);
    float*    Gr        = (float*)   alloc((size_t)MALL * H_ * 4);
    float*    Gh        = (float*)   alloc((size_t)MALL * H_ * 4);
    float*    zb        = (float*)   alloc((size_t)MALL * H_ * 4);
    _Float16* hr_h      = (_Float16*)alloc((size_t)MALL * H_ * 2);

    const int EW = 256;                               // elementwise block
    const long sNE = (long)N_ * ED_;                  // 256*64 batch stride
    const long sNN = (long)N_ * N_;

    // ---- init state + static precompute
    zero_f32<<<512,  EW, 0, stream>>>(h,  MALL * H_);
    zero_f32<<<2048, EW, 0, stream>>>(Eh, B_ * S_ * N_ * N_);
    setup_static<<<1, 256, 0, stream>>>(edge_w, edge_idx, gconv_w, w1, w2,
                                        wz_c, bz_c, wz_l, bz_l, wr_c, br_c, wr_l, br_l,
                                        wh_c, bh_c, wh_l, bh_l,
                                        A_stat, dinv_stat, AnT_h, gwT_h, w1T_h, w2T_h,
                                        wzlT_h, wrlT_h, whlT_h, pq);

    // ---- recurrent scan (T=64 sequential steps)
    for (int t = 0; t < T_; ++t) {
        int slot = t % S_;
        float inv_cnt = 1.f / (float)((t + 1) < S_ ? (t + 1) : S_);

        build_it<<<(MALL * KIT_ + EW - 1) / EW, EW, 0, stream>>>(x, h, It_h, t);
        // Y = It @ gconv_w            [4096 x 96] x [96 x 64]
        gemm_f16_wmma<<<dim3(1, MALL / 64, 1), 128, 0, stream>>>(It_h, gwT_h, Ybuf, MALL, ED_, KIT_, 0, 0, 0);
        y_to_bt<<<(MALL * ED_ + EW - 1) / EW, EW, 0, stream>>>(Ybuf, YT_h);
        // df[b] = AnStatT @ Y[b]      16 x ([256 x 256] x [256 x 64]), shared A
        gemm_f16_wmma<<<dim3(1, N_ / 64, B_), 128, 0, stream>>>(AnT_h, YT_h, dfbuf, N_, ED_, N_, 0, sNE, sNE);
        bias_to_f16<<<(MALL * ED_ + EW - 1) / EW, EW, 0, stream>>>(dfbuf, gconv_b, df_h, MALL * ED_);
        // E1 = df @ w1 ; E2 = df @ w2 [4096 x 64] x [64 x 64]
        gemm_f16_wmma<<<dim3(1, MALL / 64, 1), 128, 0, stream>>>(df_h, w1T_h, E1, MALL, ED_, ED_, 0, 0, 0);
        gemm_f16_wmma<<<dim3(1, MALL / 64, 1), 128, 0, stream>>>(df_h, w2T_h, E2, MALL, ED_, ED_, 0, 0, 0);
        epi_de<<<(MALL * ED_ + EW - 1) / EW, EW, 0, stream>>>(E1, E2, b1, b2, de1_h, de2_h);
        // M12[b] = de1[b] @ de2[b]^T  16 x ([256 x 64] x [64 x 256]); Bt == de2 row-major
        gemm_f16_wmma<<<dim3(N_ / 64, N_ / 64, B_), 128, 0, stream>>>(de1_h, de2_h, M12, N_, N_, ED_, sNE, sNE, sNN);
        epi_et<<<(B_ * N_ * N_ + EW - 1) / EW, EW, 0, stream>>>(M12, Eh, slot);
        // dynamic graph collapsed to matrix-vector: dinv then a = DnT * xt
        deg_dinv<<<B_, N_, 0, stream>>>(Eh, x, dinv_d, u_d, t, inv_cnt);
        compute_avec<<<B_, N_, 0, stream>>>(Eh, dinv_d, u_d, avec, inv_cnt);
        // GRU: only h @ W_bot survives as a GEMM
        f32_to_f16<<<(MALL * H_ + EW - 1) / EW, EW, 0, stream>>>(h, h_h, MALL * H_);
        gemm_f16_wmma<<<dim3(1, MALL / 64, 1), 128, 0, stream>>>(h_h, wzlT_h, Gz, MALL, H_, H_, 0, 0, 0);
        gemm_f16_wmma<<<dim3(1, MALL / 64, 1), 128, 0, stream>>>(h_h, wrlT_h, Gr, MALL, H_, H_, 0, 0, 0);
        epi_zr<<<(MALL * H_ + EW - 1) / EW, EW, 0, stream>>>(avec, pq, Gz, Gr, h, zb, hr_h);
        gemm_f16_wmma<<<dim3(1, MALL / 64, 1), 128, 0, stream>>>(hr_h, whlT_h, Gh, MALL, H_, H_, 0, 0, 0);
        epi_update_h<<<(MALL * H_ + EW - 1) / EW, EW, 0, stream>>>(avec, pq, Gh, zb, h);
    }

    // ---- classifier: [16 x 16384] @ [16384 x 4]
    classifier<<<B_ * C_, 256, 0, stream>>>(h, cls_w, cls_b, (float*)d_out);
}
